// MultiHeadAttention_20985210208374
// MI455X (gfx1250) — matrix operand order repaired and checked by reading, not compiled, not run on hardware
//
#include <hip/hip_runtime.h>
#include <hip/hip_bf16.h>
#include <cstdint>

// ---------------------------------------------------------------------------
// MHA for MI455X (gfx1250): bf16 WMMA GEMMs + flash attention (no score tensor)
// K/V tiles staged into LDS with async copies (ASYNCcnt), double-buffered.
// ---------------------------------------------------------------------------

typedef __attribute__((ext_vector_type(16))) __bf16 v16bf;
typedef __attribute__((ext_vector_type(8)))  float  v8f;
typedef __attribute__((ext_vector_type(4)))  int    i4v;

#define SEQ    4096
#define DMODEL 1024
#define NHEAD  16
#define DHEAD  64

#if defined(__has_builtin)
#if __has_builtin(__builtin_amdgcn_global_load_async_to_lds_b128)
#define HAVE_ASYNC_LDS 1
#endif
#if __has_builtin(__builtin_amdgcn_s_wait_asynccnt)
#define HAVE_WAIT_ASYNC 1
#endif
#endif

__device__ __forceinline__ uint16_t f2bf(float f) {
  uint32_t u = __float_as_uint(f);
  uint32_t r = (u + 0x7fffu + ((u >> 16) & 1u)) >> 16;   // round-to-nearest-even
  return (uint16_t)r;
}

__device__ __forceinline__ v8f wmma_bf16(v16bf a, v16bf b, v8f c) {
  // D = A(16x32 bf16) x B(32x16 bf16) + C(16x16 f32)
  return __builtin_amdgcn_wmma_f32_16x16x32_bf16(
      /*neg_a=*/false, a, /*neg_b=*/false, b,
      /*c_mod=*/(short)0, c, /*reuse_a=*/false, /*reuse_b=*/false);
}

// A fragment (16x32 bf16, this code's contiguous k order, same as ldB): lane holds row M=lane%16;
// p points at element (M, kBase + 16*half). j -> K+j (16 contiguous K values).
__device__ __forceinline__ v16bf ldA(const uint16_t* p) {
  union { v16bf v; uint4 q[2]; } u;
  u.q[0] = *reinterpret_cast<const uint4*>(p);
  u.q[1] = *reinterpret_cast<const uint4*>(p + 8);
  return u.v;
}
// B fragment (32x16 bf16): lane holds column N=lane%16; 16 contiguous K values
// starting at kBase + 16*half (contiguous-per-column layout, cf. sparse-B tables).
__device__ __forceinline__ v16bf ldB(const uint16_t* p) {
  union { v16bf v; uint4 q[2]; } u;
  u.q[0] = *reinterpret_cast<const uint4*>(p);
  u.q[1] = *reinterpret_cast<const uint4*>(p + 8);
  return u.v;
}

// One 16B global -> LDS async transfer (global_load_async_to_lds_b128).
__device__ __forceinline__ void async_cp16(const uint16_t* g, uint16_t* l) {
#ifdef HAVE_ASYNC_LDS
  typedef __attribute__((address_space(1))) i4v gi4;
  typedef __attribute__((address_space(3))) i4v li4;
  __builtin_amdgcn_global_load_async_to_lds_b128((gi4*)(void*)g, (li4*)(void*)l, 0, 0);
#else
  *reinterpret_cast<uint4*>(l) = *reinterpret_cast<const uint4*>(g);
#endif
}

// ---------------------------------------------------------------------------
// Phase 0: f32 -> bf16 conversion (vectorized)
// ---------------------------------------------------------------------------
__global__ __launch_bounds__(256) void cvt_f32_bf16(const float* __restrict__ in,
                                                    uint16_t* __restrict__ out, int n4) {
  int i = blockIdx.x * blockDim.x + threadIdx.x;
  if (i < n4) {
    float4 f = reinterpret_cast<const float4*>(in)[i];
    uint32_t lo = (uint32_t)f2bf(f.x) | ((uint32_t)f2bf(f.y) << 16);
    uint32_t hi = (uint32_t)f2bf(f.z) | ((uint32_t)f2bf(f.w) << 16);
    reinterpret_cast<uint2*>(out)[i] = make_uint2(lo, hi);
  }
}

// ---------------------------------------------------------------------------
// GEMM: C[4096x1024] = A[4096x1024] @ W^T + bias.  W is [out,in] row-major, so
// the B-matrix column n is W's row n -> contiguous b128 loads.
// 8 waves/block, each wave computes a 64x64 tile (16 wmma accumulators).
// MODE 0: f32 row-major out.  MODE 1: bf16 head-major [H][S][64].
// MODE 2: bf16 transposed [col][S]  (V^T, for contiguous attention B-frags).
// ---------------------------------------------------------------------------
template <int MODE>
__global__ __launch_bounds__(256) void gemm_bf16(const uint16_t* __restrict__ A,
                                                 const uint16_t* __restrict__ W,
                                                 const float* __restrict__ bias,
                                                 void* __restrict__ outp) {
  const int lane = threadIdx.x & 31;
  const int ln   = lane & 15;
  const int half = lane >> 4;
  const int wave = threadIdx.x >> 5;
  const int rowBase = blockIdx.y * 256 + (wave >> 1) * 64;
  const int colBase = blockIdx.x * 128 + (wave & 1) * 64;

  v8f acc[4][4] = {};
  const uint16_t* Arow[4];
  const uint16_t* Wrow[4];
#pragma unroll
  for (int i = 0; i < 4; ++i)
    Arow[i] = A + (size_t)(rowBase + i * 16 + ln) * DMODEL + 16 * half;
#pragma unroll
  for (int j = 0; j < 4; ++j)
    Wrow[j] = W + (size_t)(colBase + j * 16 + ln) * DMODEL + 16 * half;

  for (int k0 = 0; k0 < DMODEL; k0 += 32) {
    if (k0 + 32 < DMODEL) {                     // global_prefetch_b8 next chunk
      __builtin_prefetch(Arow[0] + k0 + 32, 0, 0);
      __builtin_prefetch(Wrow[0] + k0 + 32, 0, 0);
    }
    v16bf a[4], b[4];
#pragma unroll
    for (int i = 0; i < 4; ++i) a[i] = ldA(Arow[i] + k0);
#pragma unroll
    for (int j = 0; j < 4; ++j) b[j] = ldB(Wrow[j] + k0);
#pragma unroll
    for (int i = 0; i < 4; ++i)
#pragma unroll
      for (int j = 0; j < 4; ++j)
        acc[i][j] = wmma_bf16(a[i], b[j], acc[i][j]);
  }

#pragma unroll
  for (int j = 0; j < 4; ++j) {
    const int col = colBase + j * 16 + ln;
    const float bj = bias[col];
#pragma unroll
    for (int i = 0; i < 4; ++i) {
#pragma unroll
      for (int r = 0; r < 8; ++r) {
        const int row = rowBase + i * 16 + r + 8 * half;   // C layout: VGPR r, half
        const float v = acc[i][j][r] + bj;
        if (MODE == 0) {
          reinterpret_cast<float*>(outp)[(size_t)row * DMODEL + col] = v;
        } else if (MODE == 1) {
          reinterpret_cast<uint16_t*>(outp)[(size_t)(col >> 6) * SEQ * DHEAD +
                                            (size_t)row * DHEAD + (col & 63)] = f2bf(v);
        } else {
          reinterpret_cast<uint16_t*>(outp)[(size_t)col * SEQ + row] = f2bf(v);
        }
      }
    }
  }
}

// ---------------------------------------------------------------------------
// Flash attention: block = 8 waves, one head, 128 query rows (16 per wave).
// K/V 64x64 tiles are staged once per block into LDS (async, double-buffered);
// P goes through per-wave LDS to re-layout C->A.  V is pre-transposed so both
// staging and B-frag reads are contiguous 16B accesses.
// ---------------------------------------------------------------------------
__global__ __launch_bounds__(256) void flash_attn(const uint16_t* __restrict__ qh,
                                                  const uint16_t* __restrict__ kh,
                                                  const uint16_t* __restrict__ vT,
                                                  uint16_t* __restrict__ outb) {
  __shared__ __align__(16) uint16_t kTile[2][64][DHEAD];   // [key][d]   2 x 8 KB
  __shared__ __align__(16) uint16_t vTile[2][DHEAD][64];   // [d][key]   2 x 8 KB
  __shared__ __align__(16) uint16_t ldsP[8][16][DHEAD];    // per-wave P   16 KB

  const int tid  = threadIdx.x;
  const int lane = tid & 31;
  const int ln   = lane & 15;
  const int half = lane >> 4;
  const int wave = tid >> 5;
  const int h     = blockIdx.x;
  const int qBase = blockIdx.y * 128 + wave * 16;

  const uint16_t* qptr = qh + (size_t)h * SEQ * DHEAD;
  const uint16_t* kptr = kh + (size_t)h * SEQ * DHEAD;
  const uint16_t* vptr = vT + (size_t)h * DHEAD * SEQ;

  // Stage one 64-key K tile and V^T tile into LDS buffer `buf`.
  // Chunk c (16B = 8 elems): K linear (tile is contiguous in global);
  // V^T chunk c -> d = c/8, key-chunk = c%8.  4 transfers per thread.
  auto stage = [&](int kt, int buf) {
    const uint16_t* kg0 = kptr + (size_t)kt * DHEAD + tid * 8;
    const uint16_t* vg0 = vptr + (size_t)(tid >> 3) * SEQ + kt + (tid & 7) * 8;
    const uint16_t* vg1 = vptr + (size_t)(32 + (tid >> 3)) * SEQ + kt + (tid & 7) * 8;
    uint16_t* kl = &kTile[buf][0][0] + tid * 8;
    uint16_t* vl = &vTile[buf][0][0] + tid * 8;
    async_cp16(kg0,        kl);
    async_cp16(kg0 + 2048, kl + 2048);
    async_cp16(vg0,        vl);
    async_cp16(vg1,        vl + 2048);
  };

  v16bf qf[2];
#pragma unroll
  for (int c = 0; c < 2; ++c)
    qf[c] = ldA(qptr + (size_t)(qBase + ln) * DHEAD + c * 32 + 16 * half);

  float m[8], l[8];
  v8f o[4] = {};
#pragma unroll
  for (int r = 0; r < 8; ++r) { m[r] = -1e30f; l[r] = 0.f; }

  stage(0, 0);
  for (int kt = 0, it = 0; kt < SEQ; kt += 64, ++it) {
    const int buf = it & 1;
    __syncthreads();                      // WAR: all waves done reading buf^1
    const bool more = (kt + 64 < SEQ);
    if (more) stage(kt + 64, buf ^ 1);    // overlap next tile with this compute
#ifdef HAVE_ASYNC_LDS
    // current buf's 4 transfers done; next buf's 4 may remain in flight
#ifdef HAVE_WAIT_ASYNC
    if (more) __builtin_amdgcn_s_wait_asynccnt(4);
    else      __builtin_amdgcn_s_wait_asynccnt(0);
#else
    if (more) asm volatile("s_wait_asynccnt 4" ::: "memory");
    else      asm volatile("s_wait_asynccnt 0" ::: "memory");
#endif
#endif
    __syncthreads();                      // current tiles visible block-wide

    // ---- S = (Q @ K^T) * 1/sqrt(d) ----
    v8f s[4] = {};
#pragma unroll
    for (int c = 0; c < 2; ++c)
#pragma unroll
      for (int nt = 0; nt < 4; ++nt) {
        v16bf kf = ldB(&kTile[buf][nt * 16 + ln][c * 32 + 16 * half]);
        s[nt] = wmma_bf16(qf[c], kf, s[nt]);
      }
#pragma unroll
    for (int nt = 0; nt < 4; ++nt)
#pragma unroll
      for (int r = 0; r < 8; ++r) s[nt][r] *= 0.125f;   // 1/sqrt(64)

    // ---- online softmax: row max / rescale / row sum (16-lane half-groups) ----
    float al[8];
#pragma unroll
    for (int r = 0; r < 8; ++r) {
      float t = fmaxf(fmaxf(s[0][r], s[1][r]), fmaxf(s[2][r], s[3][r]));
#pragma unroll
      for (int off = 1; off < 16; off <<= 1) t = fmaxf(t, __shfl_xor(t, off, 32));
      const float mn = fmaxf(m[r], t);
      al[r] = __expf(m[r] - mn);
      m[r] = mn;
    }
#pragma unroll
    for (int r = 0; r < 8; ++r) {
      float ps = 0.f;
#pragma unroll
      for (int nt = 0; nt < 4; ++nt) {
        const float p = __expf(s[nt][r] - m[r]);
        s[nt][r] = p;
        ps += p;
      }
#pragma unroll
      for (int off = 1; off < 16; off <<= 1) ps += __shfl_xor(ps, off, 32);
      l[r] = l[r] * al[r] + ps;
    }

    // ---- P: C-layout -> row-major LDS -> A-layout (LDS in-order per wave) ----
#pragma unroll
    for (int nt = 0; nt < 4; ++nt)
#pragma unroll
      for (int r = 0; r < 8; ++r)
        ldsP[wave][r + 8 * half][nt * 16 + ln] = f2bf(s[nt][r]);
    asm volatile("s_wait_dscnt 0" ::: "memory");

#pragma unroll
    for (int nt = 0; nt < 4; ++nt)
#pragma unroll
      for (int r = 0; r < 8; ++r) o[nt][r] *= al[r];

    // ---- O += P @ V ----
#pragma unroll
    for (int c = 0; c < 2; ++c) {
      v16bf pf = ldA(&ldsP[wave][ln][c * 32 + 16 * half]);
#pragma unroll
      for (int nt = 0; nt < 4; ++nt) {
        v16bf vf = ldB(&vTile[buf][nt * 16 + ln][c * 32 + 16 * half]);
        o[nt] = wmma_bf16(pf, vf, o[nt]);
      }
    }
  }

  // ---- normalize and store bf16, row-major [S][DMODEL] ----
#pragma unroll
  for (int r = 0; r < 8; ++r) {
    const float inv = 1.f / l[r];
    const int row = qBase + r + 8 * half;
#pragma unroll
    for (int nt = 0; nt < 4; ++nt)
      outb[(size_t)row * DMODEL + h * DHEAD + nt * 16 + ln] = f2bf(o[nt][r] * inv);
  }
}

// ---------------------------------------------------------------------------
// Launch
// ---------------------------------------------------------------------------
extern "C" void kernel_launch(void* const* d_in, const int* in_sizes, int n_in,
                              void* d_out, int out_size, void* d_ws, size_t ws_size,
                              hipStream_t stream) {
  const float* x  = (const float*)d_in[0];
  const float* wq = (const float*)d_in[1];
  const float* bq = (const float*)d_in[2];
  const float* wk = (const float*)d_in[3];
  const float* bk = (const float*)d_in[4];
  const float* wv = (const float*)d_in[5];
  const float* bv = (const float*)d_in[6];
  const float* wo = (const float*)d_in[7];
  const float* bo = (const float*)d_in[8];

  char* ws = (char*)d_ws;                       // 48 MiB used
  uint16_t* xb  = (uint16_t*)(ws);              //  8 MiB  x  (bf16)
  uint16_t* wqb = (uint16_t*)(ws + (8ull  << 20));
  uint16_t* wkb = (uint16_t*)(ws + (10ull << 20));
  uint16_t* wvb = (uint16_t*)(ws + (12ull << 20));
  uint16_t* wob = (uint16_t*)(ws + (14ull << 20));
  uint16_t* qb  = (uint16_t*)(ws + (16ull << 20));  // [H][S][64]
  uint16_t* kb  = (uint16_t*)(ws + (24ull << 20));  // [H][S][64]
  uint16_t* vbt = (uint16_t*)(ws + (32ull << 20));  // [H][64][S]  (V^T)
  uint16_t* ob  = (uint16_t*)(ws + (40ull << 20));  // [S][DMODEL]

  const int nX = SEQ * DMODEL, nW = DMODEL * DMODEL;
  cvt_f32_bf16<<<nX / 1024, 256, 0, stream>>>(x,  xb,  nX / 4);
  cvt_f32_bf16<<<nW / 1024, 256, 0, stream>>>(wq, wqb, nW / 4);
  cvt_f32_bf16<<<nW / 1024, 256, 0, stream>>>(wk, wkb, nW / 4);
  cvt_f32_bf16<<<nW / 1024, 256, 0, stream>>>(wv, wvb, nW / 4);
  cvt_f32_bf16<<<nW / 1024, 256, 0, stream>>>(wo, wob, nW / 4);

  dim3 gg(DMODEL / 128, SEQ / 256);
  gemm_bf16<1><<<gg, 256, 0, stream>>>(xb, wqb, bq, qb);
  gemm_bf16<1><<<gg, 256, 0, stream>>>(xb, wkb, bk, kb);
  gemm_bf16<2><<<gg, 256, 0, stream>>>(xb, wvb, bv, vbt);

  flash_attn<<<dim3(NHEAD, SEQ / 128), 256, 0, stream>>>(qb, kb, vbt, ob);

  gemm_bf16<0><<<gg, 256, 0, stream>>>(ob, wob, bo, d_out);
}
